// GCNMultiScal_77884936946097
// MI455X (gfx1250) — matrix operand order repaired and checked by reading, not compiled, not run on hardware
//
#include <hip/hip_runtime.h>

// ---------- CDNA5 types ----------
typedef __attribute__((ext_vector_type(16))) __bf16        v16bf;
typedef __attribute__((ext_vector_type(8)))  float         v8f;
typedef __attribute__((ext_vector_type(4)))  float         f32x4;
typedef __attribute__((ext_vector_type(4)))  unsigned int  u32x4;
typedef __attribute__((ext_vector_type(2)))  unsigned int  u32x2;

#define TM 128      // rows per block
#define TN 128      // cols per block (= feature dim F)
#define TK 32       // K step (bf16 WMMA K)
#define PA 40       // LDS pitch (bf16 elems): 80B -> 16B aligned, bank-spread
#define PB 40
#define F  128
#define NNODES 8192

union V16U { u32x4 q[2]; v16bf v; };

// split f32 x ~= hi + lo  (two bf16; lo captures mantissa bits 9..16)
__device__ inline void split2(float x, unsigned short& h, unsigned short& l) {
    unsigned u = __float_as_uint(x);
    h = (unsigned short)(u >> 16);
    float hf = __uint_as_float(u & 0xFFFF0000u);
    float rem = x - hf;
    l = (unsigned short)(__float_as_uint(rem) >> 16);
}

// Generic   out[b] = op( A[b] (MxK) @ B[b] (KxF) )   with optional bias+ReLU.
// A row-major (lda), B row-major pitch F. Per-batch strides may be 0 (shared operand).
__global__ __launch_bounds__(256)
void gcn_gemm_bf16x3(const float* __restrict__ A, long long strideA, int lda,
                     const float* __restrict__ B, long long strideB,
                     const float* __restrict__ bias,
                     float* __restrict__ out, long long strideO,
                     int K, int relu)
{
    __shared__ unsigned short AsH[TM * PA];
    __shared__ unsigned short AsL[TM * PA];
    __shared__ unsigned short BtH[TN * PB];   // B transposed: Bt[f][k]
    __shared__ unsigned short BtL[TN * PB];

    const int tid  = threadIdx.x;
    const int lane = tid & 31;
    const int wv   = tid >> 5;        // 8 waves
    const int n    = lane & 15;       // col within 16-tile / row within A-frag
    const int hh   = lane >> 4;       // lane half (WMMA K-half select)

    // wave tiling: 4 row-groups x 2 col-groups -> each wave owns 32 rows x 64 cols
    const int wr = wv & 3;            // row group: rows [32*wr, 32*wr+32)
    const int wc = wv >> 2;           // col group: cols [64*wc, 64*wc+64)

    const int batch = blockIdx.y;
    const int row0  = blockIdx.x * TM;

    const float* Ab = A + (long long)batch * strideA + (long long)row0 * lda;
    const float* Bb = B + (long long)batch * strideB;

    v8f acc[2][4];                    // [row half ri][col tile j]
    #pragma unroll
    for (int ri = 0; ri < 2; ++ri)
        #pragma unroll
        for (int j = 0; j < 4; ++j)
            #pragma unroll
            for (int r = 0; r < 8; ++r) acc[ri][j][r] = 0.0f;

    // staging roles
    const int ar  = tid >> 1;         // A: row 0..127
    const int ac0 = (tid & 1) * 16;   // A: 16-float half of the 32-wide K slab
    const int bk  = tid >> 3;         // B: k row 0..31
    const int fb  = (tid & 7) * 16;   // B: 16-col slab

    // register double buffer for global tiles (prefetch next while computing)
    f32x4 pva[4], pvb[4];
    #pragma unroll
    for (int q = 0; q < 4; ++q) {
        pva[q] = *(const f32x4*)(Ab + (long long)ar * lda + ac0 + q * 4);
        pvb[q] = *(const f32x4*)(Bb + (long long)bk * F + fb + q * 4);
    }

    for (int k0 = 0; k0 < K; k0 += TK) {
        // ---- drain prefetch regs -> hi/lo bf16 LDS tiles
        #pragma unroll
        for (int q = 0; q < 4; ++q) {
            unsigned short h0,l0,h1,l1,h2,l2,h3,l3;
            split2(pva[q][0],h0,l0); split2(pva[q][1],h1,l1);
            split2(pva[q][2],h2,l2); split2(pva[q][3],h3,l3);
            u32x2 ph, pl;
            ph[0] = (unsigned)h0 | ((unsigned)h1 << 16);
            ph[1] = (unsigned)h2 | ((unsigned)h3 << 16);
            pl[0] = (unsigned)l0 | ((unsigned)l1 << 16);
            pl[1] = (unsigned)l2 | ((unsigned)l3 << 16);
            *(u32x2*)&AsH[ar * PA + ac0 + q * 4] = ph;
            *(u32x2*)&AsL[ar * PA + ac0 + q * 4] = pl;
        }
        #pragma unroll
        for (int q = 0; q < 4; ++q) {
            #pragma unroll
            for (int e = 0; e < 4; ++e) {
                unsigned short h, l;
                split2(pvb[q][e], h, l);
                BtH[(fb + q * 4 + e) * PB + bk] = h;
                BtL[(fb + q * 4 + e) * PB + bk] = l;
            }
        }
        __syncthreads();

        // ---- issue next tile's global loads; latency overlaps the WMMA burst
        if (k0 + TK < K) {
            const int kn = k0 + TK;
            #pragma unroll
            for (int q = 0; q < 4; ++q) {
                pva[q] = *(const f32x4*)(Ab + (long long)ar * lda + kn + ac0 + q * 4);
                pvb[q] = *(const f32x4*)(Bb + (long long)(kn + bk) * F + fb + q * 4);
            }
        }

        // ---- A fragments (two 16x32 row halves of this wave's 32 rows)
        V16U aH[2], aL[2];
        #pragma unroll
        for (int ri = 0; ri < 2; ++ri) {
            const int abase = (wr * 32 + ri * 16 + n) * PA + hh * 8;
            aH[ri].q[0] = *(const u32x4*)&AsH[abase];
            aH[ri].q[1] = *(const u32x4*)&AsH[abase + 16];
            aL[ri].q[0] = *(const u32x4*)&AsL[abase];
            aL[ri].q[1] = *(const u32x4*)&AsL[abase + 16];
        }

        // ---- 4 column tiles (this wave's 64 cols), B frag reused across 2 row halves
        #pragma unroll
        for (int j = 0; j < 4; ++j) {
            V16U bH, bL;
            const int bbase = (wc * 64 + j * 16 + n) * PB + hh * 8;
            bH.q[0] = *(const u32x4*)&BtH[bbase];
            bH.q[1] = *(const u32x4*)&BtH[bbase + 16];
            bL.q[0] = *(const u32x4*)&BtL[bbase];
            bL.q[1] = *(const u32x4*)&BtL[bbase + 16];
            #pragma unroll
            for (int ri = 0; ri < 2; ++ri) {
                // bf16x3 split-precision accumulate in f32
                acc[ri][j] = __builtin_amdgcn_wmma_f32_16x16x32_bf16(false, aH[ri].v, false, bH.v, (short)0, acc[ri][j], false, false);
                acc[ri][j] = __builtin_amdgcn_wmma_f32_16x16x32_bf16(false, aH[ri].v, false, bL.v, (short)0, acc[ri][j], false, false);
                acc[ri][j] = __builtin_amdgcn_wmma_f32_16x16x32_bf16(false, aL[ri].v, false, bH.v, (short)0, acc[ri][j], false, false);
            }
        }
        __syncthreads();
    }

    // ---- epilogue: optional bias + ReLU, f32 stores
    // C/D layout: VGPR r holds row (r + 8*laneHalf), lane n = col
    #pragma unroll
    for (int ri = 0; ri < 2; ++ri) {
        float* Ob = out + (long long)batch * strideO
                  + (long long)(row0 + wr * 32 + ri * 16) * F;
        #pragma unroll
        for (int j = 0; j < 4; ++j) {
            const int f = wc * 64 + j * 16 + n;
            float bj = bias ? bias[f] : 0.0f;
            #pragma unroll
            for (int r = 0; r < 8; ++r) {
                const int row = r + hh * 8;
                float v = acc[ri][j][r] + bj;
                if (relu) v = fmaxf(v, 0.0f);
                Ob[(long long)row * F + f] = v;
            }
        }
    }
}

extern "C" void kernel_launch(void* const* d_in, const int* in_sizes, int n_in,
                              void* d_out, int out_size, void* d_ws, size_t ws_size,
                              hipStream_t stream)
{
    (void)in_sizes; (void)n_in; (void)out_size; (void)ws_size;
    const float* x = (const float*)d_in[0];
    const float* L[4] = {(const float*)d_in[1], (const float*)d_in[2],
                         (const float*)d_in[3], (const float*)d_in[4]};
    const float* W[4][2];
    const float* bs[4][2];
    int idx = 5;
    for (int b = 0; b < 4; ++b)
        for (int l = 0; l < 2; ++l) {
            W[b][l]  = (const float*)d_in[idx++];
            bs[b][l] = (const float*)d_in[idx++];
        }

    const long long NF  = (long long)NNODES * F;   // per-batch matrix
    const long long BNF = 4LL * NF;                // one [B,N,F] tensor

    float* S = (float*)d_ws;        // support buffer  [B,N,F]
    float* H = S + BNF;             // hidden buffer   [B,N,F]
    float* O = (float*)d_out;

    dim3 grid(NNODES / TM, 4), blk(256);
    for (int b = 0; b < 4; ++b) {
        // layer 0: S = L@x ; H = ReLU(S@W0 + b0)
        gcn_gemm_bf16x3<<<grid, blk, 0, stream>>>(L[b], 0, NNODES, x, NF, nullptr,
                                                  S, NF, NNODES, 0);
        gcn_gemm_bf16x3<<<grid, blk, 0, stream>>>(S, NF, F, W[b][0], 0, bs[b][0],
                                                  H, NF, F, 1);
        // layer 1: S = L@H ; out_b = ReLU(S@W1 + b1)
        gcn_gemm_bf16x3<<<grid, blk, 0, stream>>>(L[b], 0, NNODES, H, NF, nullptr,
                                                  S, NF, NNODES, 0);
        gcn_gemm_bf16x3<<<grid, blk, 0, stream>>>(S, NF, F, W[b][1], 0, bs[b][1],
                                                  O + (long long)b * BNF, NF, F, 1);
    }
}
